// GRUCellNew_36043365548431
// MI455X (gfx1250) — matrix operand order repaired and checked by reading, not compiled, not run on hardware
//
#include <hip/hip_runtime.h>

// GRU single step, B=131072, I=H=256, 3H=768.
// Strategy: f32->f16 convert, v_wmma_f32_16x16x32_f16 GEMMs, fused gate epilogue.
// x/h streamed once (non-temporal); weights converted to f16 once into d_ws and
// served from L2/WGP$; A-fragments register-resident across the column loop.

typedef __attribute__((ext_vector_type(16))) _Float16 v16h;
typedef __attribute__((ext_vector_type(8)))  _Float16 v8h;
typedef __attribute__((ext_vector_type(8)))  float    v8f;
typedef __attribute__((ext_vector_type(4)))  float    v4f;

#define IDIM 256
#define HDIM 256
#define GDIM 768   // 3*H

// ---- weight f32 -> f16 conversion (runs every launch; 1.5 MB read) ----
__global__ void cvt_weights(const float* __restrict__ wih,
                            const float* __restrict__ whh,
                            _Float16* __restrict__ w16, int nw) {
  int i = blockIdx.x * blockDim.x + threadIdx.x;
  if (i < nw)            w16[i] = (_Float16)wih[i];
  else if (i < 2 * nw)   w16[i] = (_Float16)whh[i - nw];
}

// Load one 16-bit B fragment (32x16, K x N) for V_WMMA_F32_16X16X32_F16.
// Layout used here (contiguous k order, same as A): lanes 0-15 hold col n=lane with K=k0..k0+15 contiguous (8 VGPRs,
// 2 halves each); lanes 16-31 hold K=k0+16..k0+31.  B[k][n] = W[c=n][k], and W
// rows are contiguous in k, so each lane reads 16 contiguous halves of a W row.
__device__ __forceinline__ v16h load_bfrag(const _Float16* __restrict__ w,
                                           int crow, int k0) {
  union { v16h v; v8h p[2]; } f;
  const v8h* src = (const v8h*)(w + (size_t)crow * IDIM + k0);
  f.p[0] = src[0];
  f.p[1] = src[1];
  return f.v;
}

// Load one 16-bit A fragment (16x32, M x K) from f32 global, converting to f16.
// Layout used here (contiguous k order, same as B): lane L holds row M=L%16; lanes<16: K in [k+0,k+16);
// lanes>=16: K in [k+16,k+32). Two 32B runs per lane.
__device__ __forceinline__ v16h load_afrag_f32(const float* __restrict__ row,
                                               int kx0, int kx1) {
  union { v16h v; _Float16 h[16]; } f;
  v4f a0 = __builtin_nontemporal_load((const v4f*)(row + kx0));
  v4f a1 = __builtin_nontemporal_load((const v4f*)(row + kx0 + 4));
  v4f b0 = __builtin_nontemporal_load((const v4f*)(row + kx1));
  v4f b1 = __builtin_nontemporal_load((const v4f*)(row + kx1 + 4));
#pragma unroll
  for (int i = 0; i < 4; ++i) {
    f.h[i]      = (_Float16)a0[i];
    f.h[4 + i]  = (_Float16)a1[i];
    f.h[8 + i]  = (_Float16)b0[i];
    f.h[12 + i] = (_Float16)b1[i];
  }
  return f.v;
}

__device__ __forceinline__ float sigmoid_f(float v) {
  return 1.0f / (1.0f + __expf(-v));
}
__device__ __forceinline__ float tanh_f(float v) {
  float e2 = __expf(-2.0f * v);
  return (1.0f - e2) / (1.0f + e2);
}

// 256 threads = 8 wave32s; each wave computes a 16-row x 256-col output slab.
__global__ __launch_bounds__(256, 1)
void gru_wmma(const float* __restrict__ x, const float* __restrict__ h,
              const _Float16* __restrict__ w16, float* __restrict__ out) {
  const int wave = threadIdx.x >> 5;
  const int lane = threadIdx.x & 31;
  const int n    = lane & 15;   // column within 16-wide tile (B/C frag lane)
  const int kh   = lane >> 4;   // lane half selector
  const long rowBase = ((long)blockIdx.x * 8 + wave) * 16;

  const float* xrow = x + (rowBase + n) * (long)IDIM;  // A-frag row M = lane&15
  const float* hrow = h + (rowBase + n) * (long)HDIM;

  // Register-resident A fragments covering full K=256 for both x and h.
  v16h ax[8], ah[8];
#pragma unroll
  for (int kb = 0; kb < 8; ++kb) {
    const int kx0 = kb * 32 + kh * 16;
    const int kx1 = kb * 32 + kh * 16 + 8;
    ax[kb] = load_afrag_f32(xrow, kx0, kx1);
    ah[kb] = load_afrag_f32(hrow, kx0, kx1);
  }

  const _Float16* wih = w16;
  const _Float16* whh = w16 + (size_t)GDIM * IDIM;

#pragma unroll 1
  for (int j = 0; j < 16; ++j) {
    const int colBase = j * 16;
    v8f air = {}, aiz = {}, ain = {}, ahr = {}, ahz = {}, ahn = {};

#pragma unroll
    for (int kb = 0; kb < 8; ++kb) {
      const int k0 = kb * 32 + kh * 16;
      // igates: W_ih rows [c], [256+c], [512+c] for c = colBase + n
      v16h br = load_bfrag(wih,        colBase + n, k0);
      v16h bz = load_bfrag(wih, 256 +  colBase + n, k0);
      v16h bn = load_bfrag(wih, 512 +  colBase + n, k0);
      air = __builtin_amdgcn_wmma_f32_16x16x32_f16(false, ax[kb], false, br,
                                                   (short)0, air, false, false);
      aiz = __builtin_amdgcn_wmma_f32_16x16x32_f16(false, ax[kb], false, bz,
                                                   (short)0, aiz, false, false);
      ain = __builtin_amdgcn_wmma_f32_16x16x32_f16(false, ax[kb], false, bn,
                                                   (short)0, ain, false, false);
      // hgates: W_hh rows
      v16h cr = load_bfrag(whh,        colBase + n, k0);
      v16h cz = load_bfrag(whh, 256 +  colBase + n, k0);
      v16h cn = load_bfrag(whh, 512 +  colBase + n, k0);
      ahr = __builtin_amdgcn_wmma_f32_16x16x32_f16(false, ah[kb], false, cr,
                                                   (short)0, ahr, false, false);
      ahz = __builtin_amdgcn_wmma_f32_16x16x32_f16(false, ah[kb], false, cz,
                                                   (short)0, ahz, false, false);
      ahn = __builtin_amdgcn_wmma_f32_16x16x32_f16(false, ah[kb], false, cn,
                                                   (short)0, ahn, false, false);
    }

    // Fused gate epilogue on the C-fragment layout:
    // VGPR r, lanes 0-15 -> row M=r; lanes 16-31 -> row M=8+r; col N = lane&15.
#pragma unroll
    for (int r = 0; r < 8; ++r) {
      const long row = rowBase + r + 8 * kh;
      const float hv = h[row * (long)HDIM + colBase + n];     // L2 hit
      const float rg  = sigmoid_f(air[r] + ahr[r]);           // reset gate
      const float zg  = sigmoid_f(aiz[r] + ahz[r]);           // update gate
      const float nv  = tanh_f(ain[r] + rg * ahn[r]);         // candidate
      const float o   = nv + zg * (hv - nv);
      __builtin_nontemporal_store(o, out + row * (long)HDIM + colBase + n);
    }
  }
}

extern "C" void kernel_launch(void* const* d_in, const int* in_sizes, int n_in,
                              void* d_out, int out_size, void* d_ws, size_t ws_size,
                              hipStream_t stream) {
  const float* x   = (const float*)d_in[0];
  const float* h   = (const float*)d_in[1];
  const float* wih = (const float*)d_in[2];
  const float* whh = (const float*)d_in[3];
  float* out = (float*)d_out;

  _Float16* w16 = (_Float16*)d_ws;       // needs 2*768*256*2 = 786432 bytes
  const int nw = GDIM * IDIM;            // 196608 elements per weight matrix

  cvt_weights<<<(2 * nw + 255) / 256, 256, 0, stream>>>(wih, whh, w16, nw);

  const int B = in_sizes[0] / IDIM;      // 131072
  gru_wmma<<<B / 128, 256, 0, stream>>>(x, h, w16, out);
}
